// DynamicKGE_32126355374555
// MI455X (gfx1250) — matrix operand order repaired and checked by reading, not compiled, not run on hardware
//
#include <hip/hip_runtime.h>
#include <math.h>

#define B_TOT 8192
#define DIM   128
#define C_CTX 32
#define MPAD  48      // 33 rows padded to 3 M-tiles of 16
#define KPAD  64      // K=33 padded to 2 k-steps of 32 (bf16 WMMA K=32)

typedef __bf16 v16bf __attribute__((ext_vector_type(16)));
typedef float  v8f   __attribute__((ext_vector_type(8)));
typedef unsigned int u32x4 __attribute__((ext_vector_type(4)));
typedef int          i32x4 __attribute__((ext_vector_type(4)));
typedef int          i32x8 __attribute__((ext_vector_type(8)));

#if __has_builtin(__builtin_amdgcn_tensor_load_to_lds)
#define HAVE_TDM 1
#else
#define HAVE_TDM 0
#endif

// ---- WMMA operand loaders (CDNA5 ISA 7.12.2 layouts, wave32) ----
// A-operand (16x32 bf16): lane m = lane&15 holds row m; half=lane>>4 selects
// K chunks {k0+8h .. +7} and {k0+16+8h .. +7}.
__device__ __forceinline__ v16bf lds_a_op(const __bf16* base, int ldk, int row, int k0, int half) {
  union { v16bf v; uint4 q[2]; } u;
  const __bf16* p = base + row * ldk + k0 + half * 8;
  u.q[0] = *(const uint4*)(p);
  u.q[1] = *(const uint4*)(p + 16);
  return u.v;
}
// B-operand (32x16 bf16): lane n = lane&15 holds column n; half=lane>>4 selects
// K chunks {k0+8h .. +7} and {k0+16+8h .. +7}. Matrix stored transposed (N-major) in LDS.
__device__ __forceinline__ v16bf lds_b_op(const __bf16* base, int ldk, int col, int k0, int half) {
  union { v16bf v; uint4 q[2]; } u;
  const __bf16* p = base + col * ldk + k0 + half * 8;
  u.q[0] = *(const uint4*)(p);
  u.q[1] = *(const uint4*)(p + 16);
  return u.v;
}

__device__ __forceinline__ v8f wmma_bf16(v16bf a, v16bf b, v8f c) {
  return __builtin_amdgcn_wmma_f32_16x16x32_bf16(false, a, false, b, (short)0, c, false, false);
}

#if HAVE_TDM
// TDM 2-D tile load: global (contiguous tile, data_size=8B units) -> LDS.
// D# per CDNA5 ISA 8.3/8.4: group0 = {count=1, lds_addr, global_addr, type=2},
// group1 = {data_size=3(8B), tensor_dim0/1, tile_dim0/1, dim0_stride}.
__device__ __forceinline__ void tdm_load_2d_b8(unsigned lds_addr, unsigned long long gaddr,
                                               unsigned d0, unsigned d1, unsigned stride0) {
  u32x4 g0;
  g0.x = 1u;                                                  // count=1, user mode
  g0.y = lds_addr;                                            // bits 63:32  lds_addr
  g0.z = (unsigned)(gaddr & 0xffffffffull);                   // bits 95:64  global_addr lo
  g0.w = (unsigned)((gaddr >> 32) & 0x1ffffffull) | (2u << 30); // addr[56:32] | type=2
  i32x8 g1;
  g1[0] = (int)(3u << 16);                                    // wg_mask=0, data_size=3 (8B)
  g1[1] = (int)((d0 & 0xffffu) << 16);                        // tensor_dim0[15:0] @ bit48
  g1[2] = (int)(((d0 >> 16) & 0xffffu) | ((d1 & 0xffffu) << 16)); // dim0 hi | dim1 lo
  g1[3] = (int)(((d1 >> 16) & 0xffffu) | ((d0 & 0xffffu) << 16)); // dim1 hi | tile_dim0
  g1[4] = (int)(d1 & 0xffffu);                                // tile_dim1 (tile_dim2=0)
  g1[5] = (int)stride0;                                       // tensor_dim0_stride lo32
  g1[6] = 0;
  g1[7] = 0;
  i32x4 gz = {0, 0, 0, 0};
#if defined(__clang_major__) && (__clang_major__ >= 23)
  i32x8 gz8 = {0, 0, 0, 0, 0, 0, 0, 0};
  __builtin_amdgcn_tensor_load_to_lds(g0, g1, gz, gz, gz8, 0);
#else
  __builtin_amdgcn_tensor_load_to_lds(g0, g1, gz, gz, 0);
#endif
}
#endif

// ---- prep: W_ent/W_rel fp32 -> bf16 transposed, once per launch ----
__global__ __launch_bounds__(256)
void dynkge_prep_kernel(const float* __restrict__ W_ent, const float* __restrict__ W_rel,
                        __bf16* __restrict__ wsWT) {
  const float* W = blockIdx.y ? W_rel : W_ent;
  __bf16* dst = wsWT + (size_t)blockIdx.y * DIM * DIM;
  for (int i = threadIdx.x + blockIdx.x * 256; i < DIM * DIM; i += 256 * gridDim.x) {
    int k = i >> 7, n = i & 127;
    dst[n * DIM + k] = (__bf16)W[i];        // W^T in bf16
  }
}

__global__ __launch_bounds__(256)
void dynkge_branch_kernel(
    const float* __restrict__ entity_emb, const float* __restrict__ relation_emb,
    const float* __restrict__ entity_ctx, const float* __restrict__ relation_ctx,
    const float* __restrict__ gate_ent, const float* __restrict__ gate_rel,
    const float* __restrict__ v_ent, const float* __restrict__ v_rel,
    const int* __restrict__ pos_h, const int* __restrict__ pos_t,
    const int* __restrict__ neg_h, const int* __restrict__ neg_t,
    const int* __restrict__ pos_r, const int* __restrict__ neg_r,
    const int* __restrict__ ph_adj, const int* __restrict__ pt_adj,
    const int* __restrict__ nh_adj, const int* __restrict__ nt_adj,
    const int* __restrict__ pr_adj, const int* __restrict__ nr_adj,
    const float* __restrict__ ph_A, const float* __restrict__ pt_A,
    const float* __restrict__ nh_A, const float* __restrict__ nt_A,
    const float* __restrict__ pr_A, const float* __restrict__ nr_A,
    const __bf16* __restrict__ wsWT, float* __restrict__ ws)
{
  // K-axis permutation: ctx row c -> column c (0..31), center o -> column 32,
  // columns 33..63 zero. A columns permuted to match; rows unpermuted.
  __shared__ __align__(16) __bf16 sA[MPAD * KPAD];    // adjacency (cols permuted)
  __shared__ __align__(16) __bf16 sH0T[DIM * KPAD];   // H0^T: [n][k]
  __shared__ __align__(16) __bf16 sAH[MPAD * DIM];    // GEMM1 out, row-major
  __shared__ __align__(16) __bf16 sWT[DIM * DIM];     // W^T bf16 (TDM-filled)
  __shared__ float sO[DIM], sV[DIM], sG[DIM], sSg[DIM];
  __shared__ float sScore[MPAD], sAlpha[MPAD];

  const int b    = blockIdx.x;
  const int beta = blockIdx.y;           // 0:ph 1:pt 2:nh 3:nt 4:pr 5:nr
  const int tid  = threadIdx.x;
  const int lane = tid & 31, w = tid >> 5;
  const int half = lane >> 4, nlo = lane & 15;
  const bool isRel = (beta >= 4);

  const int* idxp; const int* adjp; const float* Ap;
  switch (beta) {
    case 0:  idxp = pos_h; adjp = ph_adj; Ap = ph_A; break;
    case 1:  idxp = pos_t; adjp = pt_adj; Ap = pt_A; break;
    case 2:  idxp = neg_h; adjp = nh_adj; Ap = nh_A; break;
    case 3:  idxp = neg_t; adjp = nt_adj; Ap = nt_A; break;
    case 4:  idxp = pos_r; adjp = pr_adj; Ap = pr_A; break;
    default: idxp = neg_r; adjp = nr_adj; Ap = nr_A; break;
  }
  const float* ctx = isRel ? relation_ctx : entity_ctx;
  const float* emb = isRel ? relation_emb : entity_emb;
  const float* gp  = isRel ? gate_rel     : gate_ent;
  const float* vp  = isRel ? v_rel        : v_ent;

  // ---- kick off W^T tile DMA (32KB, overlaps gather) ----
#if HAVE_TDM
  if (w == 0) {
    const __bf16* gsrc = wsWT + (isRel ? DIM * DIM : 0);
    // 128 rows x 128 bf16 = 128 x 32 elements of 8B, contiguous.
    tdm_load_2d_b8((unsigned)(uintptr_t)(void*)sWT,
                   (unsigned long long)(uintptr_t)gsrc,
                   DIM / 4, DIM, DIM / 4);
  }
#else
  {
    const uint4* src = (const uint4*)(wsWT + (isRel ? DIM * DIM : 0));
    uint4* dst = (uint4*)sWT;
    for (int i = tid; i < DIM * DIM * 2 / 16; i += 256) dst[i] = src[i];
  }
#endif

  // ---- phase 0: zero-fill padded LDS regions (b128 stores) ----
  {
    const uint4 z4 = {0u, 0u, 0u, 0u};
    uint4* pA = (uint4*)sA;                      // 384 x 16B
    for (int i = tid; i < MPAD * KPAD * 2 / 16; i += 256) pA[i] = z4;
    uint4* pH = (uint4*)sH0T;                    // 1024 x 16B
    for (int i = tid; i < DIM * KPAD * 2 / 16; i += 256) pH[i] = z4;
  }
  if (tid < DIM)  sSg[tid]    = 0.f;
  if (tid < MPAD) sScore[tid] = 0.f;
  __syncthreads();

  // ---- phase 1: gather + pack (fp32 -> bf16, transposed, pair-packed) ----
  const long o_idx = idxp[b];
  if (tid < DIM) {
    float ov = emb[o_idx * DIM + tid];
    sO[tid] = ov; sV[tid] = vp[tid]; sG[tid] = gp[tid];
    sH0T[tid * KPAD + C_CTX] = (__bf16)ov;       // center o -> column 32
  }
  {
    const int n = tid & 127;
    if (!isRel) {
      for (int p = (tid >> 7); p < 16; p += 2) {
        const int c0 = 2 * p, c1 = 2 * p + 1;
        long i0 = adjp[b * C_CTX + c0];
        long i1 = adjp[b * C_CTX + c1];
        union { __bf16 h[2]; unsigned u; } pk;
        pk.h[0] = (__bf16)ctx[i0 * DIM + n];
        pk.h[1] = (__bf16)ctx[i1 * DIM + n];
        *(unsigned*)&sH0T[n * KPAD + c0] = pk.u;  // even column -> aligned b32
      }
    } else {
      for (int p = (tid >> 7); p < 16; p += 2) {
        const int c0 = 2 * p, c1 = 2 * p + 1;
        long a0 = adjp[b * 2 * C_CTX + 2 * c0], a1 = adjp[b * 2 * C_CTX + 2 * c0 + 1];
        long a2 = adjp[b * 2 * C_CTX + 2 * c1], a3 = adjp[b * 2 * C_CTX + 2 * c1 + 1];
        union { __bf16 h[2]; unsigned u; } pk;
        pk.h[0] = (__bf16)(ctx[a0 * DIM + n] + ctx[a1 * DIM + n]);
        pk.h[1] = (__bf16)(ctx[a2 * DIM + n] + ctx[a3 * DIM + n]);
        *(unsigned*)&sH0T[n * KPAD + c0] = pk.u;
      }
    }
  }
  for (int i = tid; i < 33 * 33; i += 256) {
    int r = i / 33, j = i - r * 33;
    int col = (j == 0) ? C_CTX : (j - 1);        // match permuted H0 row order
    sA[r * KPAD + col] = (__bf16)Ap[(long)b * 33 * 33 + i];
  }
#if HAVE_TDM
  if (w == 0) __builtin_amdgcn_s_wait_tensorcnt(0);  // sWT DMA complete (wave 0)
#endif
  __syncthreads();

  // ---- phase 2: GEMM1  AH = A(48x64) * H0(64x128); wave w owns N-tile w ----
  const int ncol = w * 16 + nlo;
  #pragma unroll
  for (int mt = 0; mt < 3; ++mt) {
    v8f acc = {0.f, 0.f, 0.f, 0.f, 0.f, 0.f, 0.f, 0.f};
    #pragma unroll
    for (int ks = 0; ks < 2; ++ks) {
      v16bf va = lds_a_op(sA,   KPAD, mt * 16 + nlo, ks * 32, half);
      v16bf vb = lds_b_op(sH0T, KPAD, ncol,          ks * 32, half);
      acc = wmma_bf16(va, vb, acc);
    }
    const int mbase = mt * 16 + half * 8;        // C-layout: half selects M 0-7 / 8-15
    #pragma unroll
    for (int r = 0; r < 8; ++r) sAH[(mbase + r) * DIM + ncol] = (__bf16)acc[r];
  }
  __syncthreads();

  // ---- phase 3: GEMM2  H = relu(AH(48x128) * W(128x128)) ----
  v8f h[3];
  #pragma unroll
  for (int mt = 0; mt < 3; ++mt) h[mt] = (v8f){0.f, 0.f, 0.f, 0.f, 0.f, 0.f, 0.f, 0.f};
  #pragma unroll
  for (int ks = 0; ks < 4; ++ks) {
    v16bf vb = lds_b_op(sWT, DIM, ncol, ks * 32, half);
    #pragma unroll
    for (int mt = 0; mt < 3; ++mt) {
      v16bf va = lds_a_op(sAH, DIM, mt * 16 + nlo, ks * 32, half);
      h[mt] = wmma_bf16(va, vb, h[mt]);
    }
  }
  #pragma unroll
  for (int mt = 0; mt < 3; ++mt)
    #pragma unroll
    for (int r = 0; r < 8; ++r) h[mt][r] = fmaxf(h[mt][r], 0.f);

  // ---- phase 4: attention scores  s[m] = sum_n relu(H[m,n]*o[n]) * v[n] ----
  const float on = sO[ncol], vn = sV[ncol];
  #pragma unroll
  for (int mt = 0; mt < 3; ++mt) {
    float part[8];
    #pragma unroll
    for (int r = 0; r < 8; ++r) part[r] = fmaxf(h[mt][r] * on, 0.f) * vn;
    #pragma unroll
    for (int m = 1; m < 16; m <<= 1) {
      #pragma unroll
      for (int r = 0; r < 8; ++r) part[r] += __shfl_xor(part[r], m, 32);
    }
    if (nlo == 0) {
      const int mbase = mt * 16 + half * 8;
      #pragma unroll
      for (int r = 0; r < 8; ++r) atomicAdd(&sScore[mbase + r], part[r]);
    }
  }
  __syncthreads();

  // ---- phase 5: softmax over the 33 valid rows (mask padding) ----
  if (tid == 0) {
    float mx = -1e30f;
    for (int m = 0; m < 33; ++m) mx = fmaxf(mx, sScore[m]);
    float s = 0.f;
    for (int m = 0; m < 33; ++m) { float e = __expf(sScore[m] - mx); sAlpha[m] = e; s += e; }
    float inv = 1.f / s;
    for (int m = 0; m < 33; ++m)  sAlpha[m] *= inv;
    for (int m = 33; m < MPAD; ++m) sAlpha[m] = 0.f;
  }
  __syncthreads();

  // ---- phase 6: sg[n] = sum_m alpha[m] * H[m,n] ----
  {
    float part = 0.f;
    #pragma unroll
    for (int mt = 0; mt < 3; ++mt) {
      const int mbase = mt * 16 + half * 8;
      #pragma unroll
      for (int r = 0; r < 8; ++r) part += sAlpha[mbase + r] * h[mt][r];
    }
    atomicAdd(&sSg[ncol], part);
  }
  __syncthreads();

  // ---- phase 7: gated output -> workspace slot [beta][b][n] ----
  if (tid < DIM) {
    float g = 1.f / (1.f + __expf(-sG[tid]));
    ws[((long)beta * B_TOT + b) * DIM + tid] = g * sO[tid] + (1.f - g) * sSg[tid];
  }
}

__global__ __launch_bounds__(128)
void dynkge_score_kernel(const float* __restrict__ ws, float* __restrict__ out) {
  const long S = (long)B_TOT * DIM;
  const int b = blockIdx.x, tid = threadIdx.x;
  const long off = (long)b * DIM + tid;
  float ph = ws[0 * S + off], pt = ws[1 * S + off];
  float nh = ws[2 * S + off], nt = ws[3 * S + off];
  float pr = ws[4 * S + off], nr = ws[5 * S + off];
  float p = fabsf(ph + pr - pt);
  float q = fabsf(nh + nr - nt);
  #pragma unroll
  for (int m = 1; m < 32; m <<= 1) { p += __shfl_xor(p, m, 32); q += __shfl_xor(q, m, 32); }
  __shared__ float rp[4], rq[4];
  const int w = tid >> 5, lane = tid & 31;
  if (lane == 0) { rp[w] = p; rq[w] = q; }
  __syncthreads();
  if (tid == 0) {
    out[b]         = rp[0] + rp[1] + rp[2] + rp[3];
    out[B_TOT + b] = rq[0] + rq[1] + rq[2] + rq[3];
  }
}

extern "C" void kernel_launch(void* const* d_in, const int* in_sizes, int n_in,
                              void* d_out, int out_size, void* d_ws, size_t ws_size,
                              hipStream_t stream) {
  const int* pos_h  = (const int*)d_in[0];
  const int* pos_r  = (const int*)d_in[1];
  const int* pos_t  = (const int*)d_in[2];
  const int* neg_h  = (const int*)d_in[3];
  const int* neg_r  = (const int*)d_in[4];
  const int* neg_t  = (const int*)d_in[5];
  const int* ph_adj = (const int*)d_in[6];
  const int* pt_adj = (const int*)d_in[7];
  const int* nh_adj = (const int*)d_in[8];
  const int* nt_adj = (const int*)d_in[9];
  const int* pr_adj = (const int*)d_in[10];
  const int* nr_adj = (const int*)d_in[11];
  const float* ph_A = (const float*)d_in[12];
  const float* pr_A = (const float*)d_in[13];
  const float* pt_A = (const float*)d_in[14];
  const float* nh_A = (const float*)d_in[15];
  const float* nr_A = (const float*)d_in[16];
  const float* nt_A = (const float*)d_in[17];
  const float* entity_emb   = (const float*)d_in[18];
  const float* relation_emb = (const float*)d_in[19];
  const float* entity_ctx   = (const float*)d_in[20];
  const float* relation_ctx = (const float*)d_in[21];
  const float* W_ent    = (const float*)d_in[22];
  const float* W_rel    = (const float*)d_in[23];
  const float* gate_ent = (const float*)d_in[24];
  const float* gate_rel = (const float*)d_in[25];
  const float* v_ent    = (const float*)d_in[26];
  const float* v_rel    = (const float*)d_in[27];

  float* ws   = (float*)d_ws;                       // [6][B][DIM] f32 branch outputs
  __bf16* wsWT = (__bf16*)(ws + (size_t)6 * B_TOT * DIM); // + 2 x 128x128 bf16 W^T
  float* out  = (float*)d_out;                      // [2, 8192]

  dim3 pgrid(16, 2);
  dynkge_prep_kernel<<<pgrid, 256, 0, stream>>>(W_ent, W_rel, wsWT);

  dim3 grid(B_TOT, 6);
  dynkge_branch_kernel<<<grid, 256, 0, stream>>>(
      entity_emb, relation_emb, entity_ctx, relation_ctx,
      gate_ent, gate_rel, v_ent, v_rel,
      pos_h, pos_t, neg_h, neg_t, pos_r, neg_r,
      ph_adj, pt_adj, nh_adj, nt_adj, pr_adj, nr_adj,
      ph_A, pt_A, nh_A, nt_A, pr_A, nr_A, wsWT, ws);

  dynkge_score_kernel<<<B_TOT, 128, 0, stream>>>(ws, out);
}
